// GaussianSelfAttention_17471926960864
// MI455X (gfx1250) — matrix operand order repaired and checked by reading, not compiled, not run on hardware
//
#include <hip/hip_runtime.h>
#include <hip/hip_bf16.h>
#include <stdint.h>

typedef __bf16 bf16_t;
typedef __attribute__((ext_vector_type(16))) __bf16 v16bf;
typedef __attribute__((ext_vector_type(8)))  __bf16 v8bf;
typedef __attribute__((ext_vector_type(8)))  float  v8f;

#define B_   16
#define W_   32
#define H_   32
#define D_   256
#define NH_  9
#define Q_   1024      /* W*H  : queries and spatial keys        */
#define KD_  2304      /* NH*D : inner dim of output projection  */

// ---------------------------------------------------------------------------
// CDNA5 async global->LDS copy (ASYNCcnt path), ISA §15.18.3 / §10 async.
// GVS mode: 64-bit SGPR base + per-lane 32-bit byte offset; dest VGPR holds
// the 32-bit LDS byte offset (generic LDS ptr low 32 bits, per aperture rule).
// ---------------------------------------------------------------------------
__device__ __forceinline__ void async_ld_b128(uint32_t lds_off, uint32_t voff,
                                              const void* sbase) {
  asm volatile("global_load_async_to_lds_b128 %0, %1, %2"
               :
               : "v"(lds_off), "v"(voff), "s"(sbase)
               : "memory");
}

template <int N>
__device__ __forceinline__ void wait_asynccnt() {
#if __has_builtin(__builtin_amdgcn_s_wait_asynccnt)
  __builtin_amdgcn_s_wait_asynccnt((short)N);
#else
  asm volatile("s_wait_asynccnt %0" ::"i"(N) : "memory");
#endif
}

// ---------------------------------------------------------------------------
// Stage 0: per-head quadratic-form coefficients u[h][0..4]
// ---------------------------------------------------------------------------
__global__ void compute_u_kernel(const float* __restrict__ centers,
                                 const float* __restrict__ spreads,
                                 float* __restrict__ u) {
  int h = threadIdx.x;
  if (h >= NH_) return;
  float s00 = spreads[h * 4 + 0], s01 = spreads[h * 4 + 1];
  float s10 = spreads[h * 4 + 2], s11 = spreads[h * 4 + 3];
  float a = s00 * s00 + s01 * s01;      // inv_cov = S @ S^T
  float b = s00 * s10 + s01 * s11;
  float c = s10 * s10 + s11 * s11;
  float mu1 = centers[h * 2 + 0], mu2 = centers[h * 2 + 1];
  u[h * 8 + 0] = a * mu1 + b * mu2;
  u[h * 8 + 1] = c * mu2 + b * mu1;
  u[h * 8 + 2] = -0.5f * a;
  u[h * 8 + 3] = -0.5f * c;
  u[h * 8 + 4] = -b;
}

// ---------------------------------------------------------------------------
// Stage 1: scores (analytic, no R tensor) + softmax over 1024 keys -> bf16
// probs layout: [h][q][k]   (head-major so each P_h is a dense 1024x1024 A)
// ---------------------------------------------------------------------------
__global__ __launch_bounds__(256)
void scores_softmax_kernel(const float* __restrict__ u,
                           bf16_t* __restrict__ probs) {
  __shared__ float red[256];
  const int q = blockIdx.x, h = blockIdx.y;
  const int i = q >> 5, j = q & 31;
  const float u0 = u[h * 8 + 0], u1 = u[h * 8 + 1], u2 = u[h * 8 + 2];
  const float u3 = u[h * 8 + 3], u4 = u[h * 8 + 4];
  const int tid = threadIdx.x;
  const int k0 = tid * 4;

  float s[4];
#pragma unroll
  for (int t = 0; t < 4; ++t) {
    int kk = k0 + t;
    float dx = (float)((kk >> 5) - i);
    float dy = (float)((kk & 31) - j);
    s[t] = u0 * dx + u1 * dy + u2 * dx * dx + u3 * dy * dy + u4 * dx * dy;
  }
  float m = fmaxf(fmaxf(s[0], s[1]), fmaxf(s[2], s[3]));
  red[tid] = m;
  __syncthreads();
  for (int o = 128; o > 0; o >>= 1) {
    if (tid < o) red[tid] = fmaxf(red[tid], red[tid + o]);
    __syncthreads();
  }
  const float mx = red[0];
  __syncthreads();

  float e[4], sum = 0.f;
#pragma unroll
  for (int t = 0; t < 4; ++t) { e[t] = __expf(s[t] - mx); sum += e[t]; }
  red[tid] = sum;
  __syncthreads();
  for (int o = 128; o > 0; o >>= 1) {
    if (tid < o) red[tid] += red[tid + o];
    __syncthreads();
  }
  const float inv = 1.0f / red[0];

  bf16_t* p = probs + ((size_t)h * Q_ + q) * Q_ + k0;
#pragma unroll
  for (int t = 0; t < 4; ++t) p[t] = (bf16_t)(e[t] * inv);
}

// ---------------------------------------------------------------------------
// Stage 2: precision/layout converts
//   Xt[b][d][k] = bf16(hidden[b][k][d])   (B^T layout for WMMA B-fragments)
//   Wb[n][k]    = bf16(value_w[n][k])     (already [N][K])
// ---------------------------------------------------------------------------
__global__ void convert_x_kernel(const float* __restrict__ hs,
                                 bf16_t* __restrict__ Xt) {
  size_t idx = (size_t)blockIdx.x * 256 + threadIdx.x;   // b*Q*D elements
  int b = (int)(idx >> 18);
  int rem = (int)(idx & 262143);
  int k = rem >> 8, d = rem & 255;
  Xt[(size_t)b * (D_ * Q_) + (size_t)d * Q_ + k] = (bf16_t)hs[idx];
}

__global__ void convert_w_kernel(const float* __restrict__ w,
                                 bf16_t* __restrict__ Wb) {
  size_t idx = (size_t)blockIdx.x * 256 + threadIdx.x;
  Wb[idx] = (bf16_t)w[idx];
}

// ---------------------------------------------------------------------------
// Shared WMMA GEMM core: C(128x128) += A(128xK, row-major) * Bt(128xK)^T
// 8 waves, each owns a 32x64 tile = 2x4 v_wmma_f32_16x16x32_bf16 accums.
// Double-buffered LDS; tile i+1 fetched with global_load_async_to_lds_b128
// (ASYNCcnt) while tile i runs through the WMMA pipe. Async loads complete
// in order, so waiting ASYNCcnt<=4 after issuing the next tile's 4 ops
// guarantees the current tile has landed in LDS.
// ---------------------------------------------------------------------------
__device__ __forceinline__ v16bf cat8(v8bf lo, v8bf hi) {
  return __builtin_shufflevector(lo, hi, 0, 1, 2, 3, 4, 5, 6, 7,
                                 8, 9, 10, 11, 12, 13, 14, 15);
}

#define TILE_ELEMS (128 * 32)
#define TILE_BYTES (TILE_ELEMS * 2)

__device__ __forceinline__ void gemm_block_compute(
    const bf16_t* __restrict__ A, int lda,
    const bf16_t* __restrict__ Bt, int ldb,
    int K, int m0, int n0,
    bf16_t* sA, bf16_t* sB, v8f acc[2][4]) {
  const int tid  = threadIdx.x;
  const int lane = tid & 31;
  const int wv   = tid >> 5;
  const int wm   = (wv & 3) * 32;          // wave M offset in block tile
  const int wn   = (wv >> 2) * 64;         // wave N offset in block tile
  const int l15  = lane & 15;
  const int ka   = (lane < 16) ? 0 : 8;    // A frag: hi lanes take K 8..15/24..31
  const int kb   = (lane < 16) ? 0 : 8;    // B frag: hi lanes take K 8..15/24..31

  const int c0   = tid * 2;                // two 16B chunks per thread per tile
  const int lrow = c0 >> 2;                // 4 chunks per 32-elem row
  const int lcol = (c0 & 3) * 8;

  // LDS byte offsets (low 32 bits of generic LDS address = LDS offset)
  const uint32_t ldsA = (uint32_t)(uintptr_t)sA + (uint32_t)(lrow * 32 + lcol) * 2;
  const uint32_t ldsB = (uint32_t)(uintptr_t)sB + (uint32_t)(lrow * 32 + lcol) * 2;
  // per-lane global byte offsets (k=0 column of this thread's chunk)
  const uint32_t gva = (uint32_t)(((size_t)(m0 + lrow) * lda + lcol) * 2);
  const uint32_t gvb = (uint32_t)(((size_t)(n0 + lrow) * ldb + lcol) * 2);

  auto issue_tile = [&](int buf, int kk) {
    const uint32_t kbyte = (uint32_t)kk * 2;
    const uint32_t la = ldsA + (uint32_t)buf * TILE_BYTES;
    const uint32_t lb = ldsB + (uint32_t)buf * TILE_BYTES;
    async_ld_b128(la,      gva + kbyte,      A);
    async_ld_b128(la + 16, gva + kbyte + 16, A);
    async_ld_b128(lb,      gvb + kbyte,      Bt);
    async_ld_b128(lb + 16, gvb + kbyte + 16, Bt);
  };

  const int niter = K / 32;
  issue_tile(0, 0);
  for (int it = 0; it < niter; ++it) {
    const int buf = it & 1;
    if (it + 1 < niter) {
      issue_tile(buf ^ 1, (it + 1) * 32);   // prefetch next tile (async)
      wait_asynccnt<4>();                   // current tile's 4 ops retired
    } else {
      wait_asynccnt<0>();
    }
    __syncthreads();                        // publish LDS across all waves

    const bf16_t* cA = sA + buf * TILE_ELEMS;
    const bf16_t* cB = sB + buf * TILE_ELEMS;
    v16bf af[2], bfr[4];
#pragma unroll
    for (int mi = 0; mi < 2; ++mi) {
      int r = wm + mi * 16 + l15;
      v8bf lo = *(const v8bf*)(cA + r * 32 + ka);
      v8bf hi = *(const v8bf*)(cA + r * 32 + ka + 16);
      af[mi] = cat8(lo, hi);
    }
#pragma unroll
    for (int nj = 0; nj < 4; ++nj) {
      int n = wn + nj * 16 + l15;
      bfr[nj] = cat8(*(const v8bf*)(cB + n * 32 + kb), *(const v8bf*)(cB + n * 32 + kb + 16));
    }
#pragma unroll
    for (int mi = 0; mi < 2; ++mi)
#pragma unroll
      for (int nj = 0; nj < 4; ++nj)
        acc[mi][nj] = __builtin_amdgcn_wmma_f32_16x16x32_bf16(
            false, af[mi], false, bfr[nj], (short)0, acc[mi][nj], false, false);
    __syncthreads();                        // all waves done reading buf
  }
}

// ---------------------------------------------------------------------------
// GEMM1: vals[b][q][h*256+d] = sum_k P_h[q,k] * X[b,k,d]      (bf16 out)
// grid: (8 mtiles, 2 ntiles, nb*NH)
// ---------------------------------------------------------------------------
__global__ __launch_bounds__(256)
void gemm1_kernel(const bf16_t* __restrict__ probs,
                  const bf16_t* __restrict__ Xt,
                  bf16_t* __restrict__ vals, int b0) {
  __shared__ __align__(16) bf16_t sA[2 * TILE_ELEMS];
  __shared__ __align__(16) bf16_t sB[2 * TILE_ELEMS];
  const int bh = blockIdx.z;
  const int bl = bh / NH_;                 // batch within chunk
  const int h  = bh % NH_;
  const int b  = b0 + bl;
  const bf16_t* A  = probs + (size_t)h * Q_ * Q_;
  const bf16_t* Bt = Xt + (size_t)b * D_ * Q_;
  const int m0 = blockIdx.x * 128;
  const int n0 = blockIdx.y * 128;

  v8f acc[2][4] = {};
  gemm_block_compute(A, Q_, Bt, Q_, Q_, m0, n0, sA, sB, acc);

  bf16_t* C = vals + ((size_t)bl * Q_) * KD_ + (size_t)h * D_;
  const int tid = threadIdx.x, lane = tid & 31, wv = tid >> 5;
  const int wm = (wv & 3) * 32, wn = (wv >> 2) * 64;
  const int l15 = lane & 15, lh8 = (lane < 16) ? 0 : 8;
#pragma unroll
  for (int mi = 0; mi < 2; ++mi)
#pragma unroll
    for (int nj = 0; nj < 4; ++nj) {
      const int ncol = n0 + wn + nj * 16 + l15;
#pragma unroll
      for (int r = 0; r < 8; ++r) {
        const int mrow = m0 + wm + mi * 16 + lh8 + r;
        C[(size_t)mrow * KD_ + ncol] = (bf16_t)acc[mi][nj][r];
      }
    }
}

// ---------------------------------------------------------------------------
// GEMM2: out[m][n] = sum_k vals[m][k] * value_w[n][k] + bias[n]   (f32 out)
// grid: (nb*8 mtiles, 2 ntiles)
// ---------------------------------------------------------------------------
__global__ __launch_bounds__(256)
void gemm2_kernel(const bf16_t* __restrict__ vals,
                  const bf16_t* __restrict__ Wb,
                  const float* __restrict__ bias,
                  float* __restrict__ out) {
  __shared__ __align__(16) bf16_t sA[2 * TILE_ELEMS];
  __shared__ __align__(16) bf16_t sB[2 * TILE_ELEMS];
  const int m0 = blockIdx.x * 128;
  const int n0 = blockIdx.y * 128;

  v8f acc[2][4] = {};
  gemm_block_compute(vals, KD_, Wb, KD_, KD_, m0, n0, sA, sB, acc);

  const int tid = threadIdx.x, lane = tid & 31, wv = tid >> 5;
  const int wm = (wv & 3) * 32, wn = (wv >> 2) * 64;
  const int l15 = lane & 15, lh8 = (lane < 16) ? 0 : 8;
#pragma unroll
  for (int mi = 0; mi < 2; ++mi)
#pragma unroll
    for (int nj = 0; nj < 4; ++nj) {
      const int ncol = n0 + wn + nj * 16 + l15;
      const float bv = bias[ncol];
#pragma unroll
      for (int r = 0; r < 8; ++r) {
        const int mrow = m0 + wm + mi * 16 + lh8 + r;
        out[(size_t)mrow * D_ + ncol] = acc[mi][nj][r] + bv;
      }
    }
}

// ---------------------------------------------------------------------------
extern "C" void kernel_launch(void* const* d_in, const int* in_sizes, int n_in,
                              void* d_out, int out_size, void* d_ws,
                              size_t ws_size, hipStream_t stream) {
  const float* hs      = (const float*)d_in[0];   // (16,32,32,256) f32
  const float* centers = (const float*)d_in[2];   // (9,2)
  const float* spreads = (const float*)d_in[3];   // (9,2,2)
  const float* value_w = (const float*)d_in[4];   // (256, 2304)
  const float* value_b = (const float*)d_in[5];   // (256,)
  float* out = (float*)d_out;
  (void)in_sizes; (void)n_in; (void)out_size;

  char* ws = (char*)d_ws;
  size_t off = 0;
  auto alloc = [&](size_t bytes) {
    off = (off + 255) & ~(size_t)255;
    size_t p = off;
    off += bytes;
    return p;
  };
  float*  u     = (float*)(ws + alloc(NH_ * 8 * sizeof(float)));
  bf16_t* probs = (bf16_t*)(ws + alloc((size_t)NH_ * Q_ * Q_ * 2));   // 18.9 MB
  bf16_t* Xt    = (bf16_t*)(ws + alloc((size_t)B_ * D_ * Q_ * 2));    //  8.4 MB
  bf16_t* Wb    = (bf16_t*)(ws + alloc((size_t)D_ * KD_ * 2));        //  1.2 MB
  off = (off + 255) & ~(size_t)255;
  const size_t fixed = off;
  const size_t per_b = (size_t)Q_ * KD_ * 2;                          //  4.7 MB
  int cb = 1;
  if (ws_size > fixed + per_b) {
    size_t m = (ws_size - fixed) / per_b;
    cb = (int)(m > 16 ? 16 : m);
  }
  bf16_t* vals = (bf16_t*)(ws + fixed);

  compute_u_kernel<<<1, 32, 0, stream>>>(centers, spreads, u);
  scores_softmax_kernel<<<dim3(Q_, NH_), 256, 0, stream>>>(u, probs);
  convert_x_kernel<<<(B_ * Q_ * D_) / 256, 256, 0, stream>>>(hs, Xt);
  convert_w_kernel<<<(D_ * KD_) / 256, 256, 0, stream>>>(value_w, Wb);

  for (int b0 = 0; b0 < B_; b0 += cb) {
    int nb = (B_ - b0 < cb) ? (B_ - b0) : cb;
    gemm1_kernel<<<dim3(8, 2, nb * NH_), 256, 0, stream>>>(probs, Xt, vals, b0);
    gemm2_kernel<<<dim3(nb * 8, 2), 256, 0, stream>>>(
        vals, Wb, value_b, out + (size_t)b0 * Q_ * D_);
  }
}
